// KLayer_721554506345
// MI455X (gfx1250) — matrix operand order repaired and checked by reading, not compiled, not run on hardware
//
#include <hip/hip_runtime.h>
#include <hip/hip_bf16.h>

typedef __attribute__((ext_vector_type(16))) __bf16 v16bf;
typedef __attribute__((ext_vector_type(8)))  float  v8f;

#define BATCH   16
#define CH      256
#define HW      1024       // 32*32
#define PW      34         // padded spatial width/height
#define PHW     1156       // 34*34
#define KSTEPS  72         // K = 9*256 = 2304 -> 72 steps of 32

#define WMMA_BF16(A, B, C) \
  __builtin_amdgcn_wmma_f32_16x16x32_bf16(false, (A), false, (B), (short)0, (C), false, false)

// ---------------- prep kernels ----------------

__global__ void k_zero_es(float* es, int n) {
  int i = blockIdx.x * blockDim.x + threadIdx.x;
  if (i < n) es[i] = 0.f;
}

// zero a buffer of n dwords (used for padded bf16 activation buffers)
__global__ void k_zero_buf(unsigned int* p, int n) {
  int i = blockIdx.x * blockDim.x + threadIdx.x;
  if (i < n) p[i] = 0u;
}

__global__ void k_omega(const float* __restrict__ p, float* __restrict__ o) {
  int i = threadIdx.x;                    // 128 pairs
  float a = p[2*i], b = p[2*i+1];
  o[i] = sqrtf(a*a + b*b);
}

// Pack conv_w (Cout,Cin,3,3) f32 -> bf16 A-fragments.
// K order = ((kh*3+kw)*256 + ci). Layout: [mtile][kk][lane][8 dwords],
// dword j of lane L holds K pair (klocal, klocal+1) with
// klocal = 2j + (L<16 ? 0 : 16), M = mtile*16 + (L&15).
__global__ void k_pack(const float* __restrict__ w, __bf16* __restrict__ wp) {
  int id = blockIdx.x * 256 + threadIdx.x;          // 0..294911
  int j    = id & 7;
  int lane = (id >> 3) & 31;
  int rest = id >> 8;                                // mt*72 + kk, 0..1151
  int kk   = rest % KSTEPS;
  int mt   = rest / KSTEPS;
  int co   = mt*16 + (lane & 15);
  int kl   = 2*j + ((lane < 16) ? 0 : 16);
  int K    = kk*32 + kl;
  int blk  = K >> 8;                                 // kh*3+kw
  int ci   = K & 255;
  int kh   = blk / 3;
  int kw   = blk - kh*3;
  float f0 = w[(((size_t)co*CH + ci  )*3 + kh)*3 + kw];
  float f1 = w[(((size_t)co*CH + ci+1)*3 + kh)*3 + kw];
  wp[2*(size_t)id    ] = (__bf16)f0;
  wp[2*(size_t)id + 1] = (__bf16)f1;
}

// GroupNorm of c (NCHW, 64 groups of 4ch x 32x32) -> c_nhwc f32
__global__ __launch_bounds__(256) void k_gn(const float* __restrict__ c,
                                            const float* __restrict__ gw,
                                            const float* __restrict__ gb,
                                            float* __restrict__ cn) {
  __shared__ float s1[256], s2[256];
  int bg = blockIdx.x;            // 0..1023
  int b  = bg >> 6;
  int g  = bg & 63;
  int t  = threadIdx.x;
  const float* base = c + ((size_t)b*CH + g*4) * HW;   // 4096 contiguous floats
  float s = 0.f, ss = 0.f;
  for (int k = 0; k < 16; ++k) {
    float v = base[t + 256*k];
    s += v; ss += v*v;
  }
  s1[t] = s; s2[t] = ss; __syncthreads();
  for (int o = 128; o > 0; o >>= 1) {
    if (t < o) { s1[t] += s1[t+o]; s2[t] += s2[t+o]; }
    __syncthreads();
  }
  float mean = s1[0] * (1.f/4096.f);
  float var  = s2[0] * (1.f/4096.f) - mean*mean;
  float inv  = rsqrtf(var + 1e-5f);
  for (int k = 0; k < 16; ++k) {
    int e  = t + 256*k;
    int ch = e >> 10;
    int sp = e & 1023;
    int ca = g*4 + ch;
    float v = base[e];
    cn[((size_t)b*HW + sp)*CH + ca] = (v - mean) * inv * gw[ca] + gb[ca];
  }
}

// Normalize x per group of 4 channels; NCHW -> NHWC f32 + padded-NHWC bf16
__global__ __launch_bounds__(256) void k_normx(const float* __restrict__ x,
                                               float* __restrict__ xf,
                                               __bf16* __restrict__ xb) {
  int id = blockIdx.x * 256 + threadIdx.x;   // 0..1048575  (b,sp,g)
  int b  = id >> 16;
  int r  = id & 65535;
  int sp = r >> 6;
  int g  = r & 63;
  int h  = sp >> 5;
  int w  = sp & 31;
  const float* p = x + ((size_t)b*CH + g*4) * HW + sp;
  float v0 = p[0], v1 = p[HW], v2 = p[2*HW], v3 = p[3*HW];
  float nr = sqrtf(v0*v0 + v1*v1 + v2*v2 + v3*v3);
  float iv = 1.f / fmaxf(nr, 1e-12f);
  v0 *= iv; v1 *= iv; v2 *= iv; v3 *= iv;
  size_t o = ((size_t)b*HW + sp)*CH + g*4;
  xf[o] = v0; xf[o+1] = v1; xf[o+2] = v2; xf[o+3] = v3;
  size_t ob = ((size_t)b*PHW + (h+1)*PW + (w+1))*CH + g*4;
  xb[ob] = (__bf16)v0; xb[ob+1] = (__bf16)v1; xb[ob+2] = (__bf16)v2; xb[ob+3] = (__bf16)v3;
}

// ---------------- fused conv (WMMA implicit GEMM) + oscillator update ----------------

__global__ __launch_bounds__(256) void k_step(
    const __bf16* __restrict__ xbf_in, const float* __restrict__ xf_in,
    const __bf16* __restrict__ wpk,    const float* __restrict__ bias,
    const float*  __restrict__ c_n,    const float* __restrict__ omg,
    const float*  __restrict__ gamma_p,
    float* __restrict__ xf_out, __bf16* __restrict__ xbf_out,
    float* __restrict__ xs_out, float* __restrict__ es_out)
{
  const int wid  = threadIdx.x >> 5;
  const int lane = threadIdx.x & 31;
  const int gwv  = blockIdx.x * 8 + wid;     // 0..4095
  const int nt   = gwv >> 2;                 // N-tile 0..1023
  const int mg   = gwv & 3;                  // 64-channel M-group
  const int n    = nt >> 6;
  const int rem  = nt & 63;
  const int h    = rem >> 1;
  const int wcol = ((rem & 1) << 4) + (lane & 15);
  const int hs   = lane >> 4;

  v8f acc[4] = { v8f{}, v8f{}, v8f{}, v8f{} };

  // weight fragments: [mtile][kk][lane][16 bf16]; mtile stride = 72*32*16 = 36864
  const __bf16* aroot = wpk + (((size_t)(mg*4) * KSTEPS * 32 + lane) << 4);
  // padded-NHWC pixel index of the top-left tap for this lane's output column
  const int rowb = n*PHW + h*PW + wcol;

#pragma unroll 1
  for (int kh = 0; kh < 3; ++kh) {
#pragma unroll 1
    for (int kw = 0; kw < 3; ++kw) {
      const int blk = kh*3 + kw;
      const __bf16* bp = xbf_in + (size_t)(rowb + kh*PW + kw) * CH + hs*16;
      const __bf16* ap = aroot + (size_t)blk * 4096;   // 8 ksteps * 512 elems
      v16bf b  = *(const v16bf*)(bp);
      v16bf a0 = *(const v16bf*)(ap);
      v16bf a1 = *(const v16bf*)(ap + 36864);
      v16bf a2 = *(const v16bf*)(ap + 73728);
      v16bf a3 = *(const v16bf*)(ap + 110592);
#pragma unroll
      for (int kc = 0; kc < 8; ++kc) {
        v16bf bn{}, n0{}, n1{}, n2{}, n3{};
        if (kc < 7) {                       // prefetch next K-chunk (compile-time guard)
          const int e = (kc + 1) * 512;
          bn = *(const v16bf*)(bp + (kc + 1) * 32);
          n0 = *(const v16bf*)(ap + e);
          n1 = *(const v16bf*)(ap + e + 36864);
          n2 = *(const v16bf*)(ap + e + 73728);
          n3 = *(const v16bf*)(ap + e + 110592);
        }
        acc[0] = WMMA_BF16(a0, b, acc[0]);
        acc[1] = WMMA_BF16(a1, b, acc[1]);
        acc[2] = WMMA_BF16(a2, b, acc[2]);
        acc[3] = WMMA_BF16(a3, b, acc[3]);
        if (kc < 7) { b = bn; a0 = n0; a1 = n1; a2 = n2; a3 = n3; }
      }
    }
  }

  const float g    = gamma_p[0];
  const size_t pos = (size_t)n*HW + h*32 + wcol;                       // unpadded NHWC
  const size_t ppos = ((size_t)n*PHW + (h+1)*PW + (wcol+1)) * CH;      // padded NHWC
  float esl = 0.f;

#pragma unroll
  for (int rt = 0; rt < 4; ++rt) {
    const int cob = mg*64 + rt*16 + hs*8;     // 8 consecutive channels in this lane
    const float* cp = c_n  + pos*CH + cob;
    const float* xp = xf_in + pos*CH + cob;
    float x8[8], y8[8];
#pragma unroll
    for (int r = 0; r < 8; ++r) {
      x8[r] = xp[r];
      y8[r] = acc[rt][r] + bias[cob + r] + cp[r];
    }
    const float s0 = x8[0]*y8[0] + x8[1]*y8[1] + x8[2]*y8[2] + x8[3]*y8[3];
    const float s1 = x8[4]*y8[4] + x8[5]*y8[5] + x8[6]*y8[6] + x8[7]*y8[7];
    esl += s0 + s1;
    float xn[8];
#pragma unroll
    for (int r = 0; r < 8; ++r) {
      const int co   = cob + r;
      const float ow = omg[co >> 1];
      const float ox = (r & 1) ? (-ow * x8[r-1]) : (ow * x8[r+1]);
      const float s  = (r < 4) ? s0 : s1;
      xn[r] = x8[r] + g * (ox + y8[r] - s * x8[r]);
    }
    const float q0 = sqrtf(xn[0]*xn[0] + xn[1]*xn[1] + xn[2]*xn[2] + xn[3]*xn[3]);
    const float q1 = sqrtf(xn[4]*xn[4] + xn[5]*xn[5] + xn[6]*xn[6] + xn[7]*xn[7]);
    const float i0 = 1.f / fmaxf(q0, 1e-12f);
    const float i1 = 1.f / fmaxf(q1, 1e-12f);
#pragma unroll
    for (int r = 0; r < 8; ++r) {
      const float v = xn[r] * ((r < 4) ? i0 : i1);
      const int co  = cob + r;
      xf_out [pos*CH + co] = v;
      xbf_out[ppos + co]   = (__bf16)v;
      xs_out[(((size_t)n*CH + co) * HW) + h*32 + wcol] = v;   // NCHW slice
    }
  }

  float e = esl;
#pragma unroll
  for (int off = 16; off > 0; off >>= 1) e += __shfl_xor(e, off, 32);
  if (lane == 0) atomicAdd(es_out + n, -e);
}

// ---------------- host launcher ----------------

extern "C" void kernel_launch(void* const* d_in, const int* in_sizes, int n_in,
                              void* d_out, int out_size, void* d_ws, size_t ws_size,
                              hipStream_t stream) {
  (void)in_sizes; (void)n_in; (void)ws_size;
  const float* x      = (const float*)d_in[0];
  const float* c      = (const float*)d_in[1];
  const float* gamma  = (const float*)d_in[3];
  const float* conv_w = (const float*)d_in[4];
  const float* conv_b = (const float*)d_in[5];
  const float* gn_w   = (const float*)d_in[6];
  const float* gn_b   = (const float*)d_in[7];
  const float* omg_p  = (const float*)d_in[8];
  float* out = (float*)d_out;

  const int xs_elems = BATCH * CH * HW;                  // 4,194,304 per step
  const int T = (out_size - BATCH) / (xs_elems + BATCH); // out = T*xs + (T+1)*B

  char* ws = (char*)d_ws;
  float*  c_n = (float*) (ws + (size_t)0);          // 16 MiB
  float*  xf0 = (float*) (ws + (size_t)16777216);   // 16 MiB
  float*  xf1 = (float*) (ws + (size_t)33554432);   // 16 MiB
  __bf16* xb0 = (__bf16*)(ws + (size_t)50331648);   //  padded bf16, 9,469,952 B
  __bf16* xb1 = (__bf16*)(ws + (size_t)60817408);   //  padded bf16, 9,469,952 B
  __bf16* wpk = (__bf16*)(ws + (size_t)71303168);   //  1.125 MiB packed weights
  float*  omg = (float*) (ws + (size_t)72482816);   //  512 B

  float* es = out + (size_t)T * xs_elems;

  const int pad_dw = BATCH * PHW * CH / 2;          // padded buffer size in dwords

  k_zero_es <<<1, 256, 0, stream>>>(es, (T + 1) * BATCH);
  k_zero_buf<<<(pad_dw + 255) / 256, 256, 0, stream>>>((unsigned int*)xb0, pad_dw);
  k_zero_buf<<<(pad_dw + 255) / 256, 256, 0, stream>>>((unsigned int*)xb1, pad_dw);
  k_omega   <<<1, 128, 0, stream>>>(omg_p, omg);
  k_pack    <<<1152, 256, 0, stream>>>(conv_w, wpk);
  k_gn      <<<1024, 256, 0, stream>>>(c, gn_w, gn_b, c_n);
  k_normx   <<<4096, 256, 0, stream>>>(x, xf0, xb0);

  float*  xf[2] = {xf0, xf1};
  __bf16* xb[2] = {xb0, xb1};
  for (int t = 0; t < T; ++t) {
    k_step<<<512, 256, 0, stream>>>(xb[t & 1], xf[t & 1], wpk, conv_b, c_n, omg, gamma,
                                    xf[(t + 1) & 1], xb[(t + 1) & 1],
                                    out + (size_t)t * xs_elems,
                                    es + (size_t)(t + 1) * BATCH);
  }
}
